// BiaffineLabelAttention_56049323213773
// MI455X (gfx1250) — matrix operand order repaired and checked by reading, not compiled, not run on hardware
//
#include <hip/hip_runtime.h>

// Problem dims (B, S, D, L) = (4, 512, 768, 64), all f32.
constexpr int Bn = 4;
constexpr int Sn = 512;
constexpr int Dn = 768;
constexpr int Ln = 64;

typedef __attribute__((ext_vector_type(16))) __bf16 v16bf;
typedef __attribute__((ext_vector_type(8)))  float  v8f;

constexpr int    FRAG_WORDS = 2048;  // 256 fragment slots * 8 dwords
// Dynamic LDS: sU[768] + sTh[128] + sTd[128] (4 KB), A staging 32 KB,
// plus B staging 32 KB only in the no-workspace fallback.
constexpr size_t SMEM_PRE = (size_t)(Dn + 256) * 4 + 2ull * 2 * FRAG_WORDS * 4; // 36,864
constexpr size_t SMEM_ALL = (size_t)(Dn + 256) * 4 + 4ull * 2 * FRAG_WORDS * 4; // 69,632

constexpr size_t HALF_WORDS = (size_t)Bn * Sn * Dn / 2;     // dwords per bf16 array
constexpr size_t WS_NEED    = HALF_WORDS * 2 * 4;           // 6,291,456 B

// Split two f32 into packed bf16-hi / bf16-lo dwords. Truncation split:
// lo = x - hi is exact in f32; v_perm_b32 packs two high halves in one op.
static __device__ __forceinline__ void split_pair(float x0, float x1,
                                                  unsigned& hiW, unsigned& loW) {
    unsigned u0 = __builtin_bit_cast(unsigned, x0);
    unsigned u1 = __builtin_bit_cast(unsigned, x1);
    float h0f = __builtin_bit_cast(float, u0 & 0xffff0000u);
    float h1f = __builtin_bit_cast(float, u1 & 0xffff0000u);
    float l0 = x0 - h0f;
    float l1 = x1 - h1f;
    hiW = __builtin_amdgcn_perm(u1, u0, 0x07060302u);
    loW = __builtin_amdgcn_perm(__builtin_bit_cast(unsigned, l1),
                                __builtin_bit_cast(unsigned, l0), 0x07060302u);
}

struct U32x8 { uint4 a, b; };

static __device__ __forceinline__ v16bf ldFrag(const unsigned* p) {
    U32x8 t;
    t.a = *(const uint4*)p;
    t.b = *(const uint4*)(p + 4);
    return __builtin_bit_cast(v16bf, t);
}

// Prepass: elementwise bf16 hi/lo split of dep into workspace (runs once,
// removes the 256x-redundant per-block B conversion from the main loop).
__launch_bounds__(256)
__global__ void split_dep_kernel(const float* __restrict__ dep,
                                 unsigned* __restrict__ dh,
                                 unsigned* __restrict__ dl) {
    const size_t i = ((size_t)blockIdx.x * blockDim.x + threadIdx.x) * 8;
    const float4 a = *(const float4*)(dep + i);
    const float4 c = *(const float4*)(dep + i + 4);
    unsigned hw[4], lw[4];
    split_pair(a.x, a.y, hw[0], lw[0]);
    split_pair(a.z, a.w, hw[1], lw[1]);
    split_pair(c.x, c.y, hw[2], lw[2]);
    split_pair(c.z, c.w, hw[3], lw[3]);
    *(uint4*)(dh + i / 2) = make_uint4(hw[0], hw[1], hw[2], hw[3]);
    *(uint4*)(dl + i / 2) = make_uint4(lw[0], lw[1], lw[2], lw[3]);
}

// Block tile 128x128, 8 waves, wave tile 64x32 (4x2 WMMA tiles).
// PREB=true: B (dep) hi/lo read from precomputed global bf16; producers
// convert only A (head * U[l]) into LDS fragments.
template <bool PREB>
__launch_bounds__(256)
__global__ void biaffine_wmma_kernel(const float* __restrict__ head,
                                     const float* __restrict__ dep,
                                     const float* __restrict__ U,     // (L, D)
                                     const float* __restrict__ W,     // (L, 2D)
                                     const float* __restrict__ bias,  // (L,)
                                     float* __restrict__ out,         // (B, L, S, S)
                                     const unsigned* __restrict__ dh,
                                     const unsigned* __restrict__ dl) {
    extern __shared__ unsigned char smemRaw[];
    float*    sU   = (float*)smemRaw;
    float*    sTh  = sU + Dn;
    float*    sTd  = sTh + 128;
    unsigned* sAhi = (unsigned*)(sTd + 128);
    unsigned* sAlo = sAhi + 2 * FRAG_WORDS;
    unsigned* sBhi = sAlo + 2 * FRAG_WORDS;  // used only when !PREB
    unsigned* sBlo = sBhi + 2 * FRAG_WORDS;  // used only when !PREB

    const int z = blockIdx.z;  // b*L + l
    const int b = z / Ln;
    const int l = z - b * Ln;
    const int mBlock = blockIdx.y * 128;
    const int nBlock = blockIdx.x * 128;
    const int tid = threadIdx.x;

    // ---- Stage U[l,:] ----
    const float* Urow = U + (size_t)l * Dn;
    for (int i = tid; i < Dn; i += 256) sU[i] = Urow[i];

    // ---- t2_h (128 rows) / t2_d (128 cols): one exact f32 dot per thread ----
    const float* Wh = W + (size_t)l * (2 * Dn);
    const float* Wd = Wh + Dn;
    {
        const bool isRow = tid < 128;
        const int  idx   = isRow ? tid : tid - 128;
        const float* src = isRow ? head + ((size_t)(b * Sn + mBlock + idx)) * Dn
                                 : dep  + ((size_t)(b * Sn + nBlock + idx)) * Dn;
        const float* w   = isRow ? Wh : Wd;
        float s = 0.f;
        for (int d = 0; d < Dn; d += 4) {
            float4 hv = *(const float4*)(src + d);
            float4 wv = *(const float4*)(w + d);
            s += hv.x * wv.x + hv.y * wv.y + hv.z * wv.z + hv.w * wv.w;
        }
        if (isRow) sTh[idx] = s; else sTd[idx] = s;
    }

    // ---- Producer geometry (fragment slot = (tile, lane) = (tid>>5, tid&31)) ----
    // A (16-bit 16x32): lanes 0-15 K 0..15; lanes 16-31 K 16..31 (contiguous 16).
    // B (16-bit 32x16): lanes 0-15 K 0..15; lanes 16-31 K 16..31 (contiguous 16).
    const int pt  = tid >> 5;
    const int pl  = tid & 31;
    const int plh = pl & 15;
    const int pdA = (pl >= 16) ? 16 : 0;
    const int pdB = (pl >= 16) ? 16 : 0;
    const float* aProd = head + ((size_t)(b * Sn + mBlock + pt * 16 + plh)) * Dn;
    const float* bProd = dep  + ((size_t)(b * Sn + nBlock + pt * 16 + plh)) * Dn;

    auto produce = [&](int kk, int buf) {
        unsigned hiW[8], loW[8];
        // A slab: scale by U, hi/lo split into LDS fragments.
        #pragma unroll
        for (int c = 0; c < 2; ++c) {
            const int kb = kk + pdA + c * 8;
            const float4 h0 = *(const float4*)(aProd + kb);
            const float4 h1 = *(const float4*)(aProd + kb + 4);
            const float4 u0 = *(const float4*)(sU + kb);
            const float4 u1 = *(const float4*)(sU + kb + 4);
            split_pair(h0.x * u0.x, h0.y * u0.y, hiW[c * 4 + 0], loW[c * 4 + 0]);
            split_pair(h0.z * u0.z, h0.w * u0.w, hiW[c * 4 + 1], loW[c * 4 + 1]);
            split_pair(h1.x * u1.x, h1.y * u1.y, hiW[c * 4 + 2], loW[c * 4 + 2]);
            split_pair(h1.z * u1.z, h1.w * u1.w, hiW[c * 4 + 3], loW[c * 4 + 3]);
        }
        unsigned* dA  = sAhi + buf * FRAG_WORDS + tid * 8;
        unsigned* dAl = sAlo + buf * FRAG_WORDS + tid * 8;
        *(uint4*)(dA)      = make_uint4(hiW[0], hiW[1], hiW[2], hiW[3]);
        *(uint4*)(dA + 4)  = make_uint4(hiW[4], hiW[5], hiW[6], hiW[7]);
        *(uint4*)(dAl)     = make_uint4(loW[0], loW[1], loW[2], loW[3]);
        *(uint4*)(dAl + 4) = make_uint4(loW[4], loW[5], loW[6], loW[7]);
        if constexpr (!PREB) {
            // Fallback: convert B slab into LDS as well.
            #pragma unroll
            for (int q = 0; q < 2; ++q) {
                const int kb = kk + pdB + q * 8;
                const float4 d0 = *(const float4*)(bProd + kb);
                const float4 d1 = *(const float4*)(bProd + kb + 4);
                split_pair(d0.x, d0.y, hiW[q * 4 + 0], loW[q * 4 + 0]);
                split_pair(d0.z, d0.w, hiW[q * 4 + 1], loW[q * 4 + 1]);
                split_pair(d1.x, d1.y, hiW[q * 4 + 2], loW[q * 4 + 2]);
                split_pair(d1.z, d1.w, hiW[q * 4 + 3], loW[q * 4 + 3]);
            }
            unsigned* dB  = sBhi + buf * FRAG_WORDS + tid * 8;
            unsigned* dBl = sBlo + buf * FRAG_WORDS + tid * 8;
            *(uint4*)(dB)      = make_uint4(hiW[0], hiW[1], hiW[2], hiW[3]);
            *(uint4*)(dB + 4)  = make_uint4(hiW[4], hiW[5], hiW[6], hiW[7]);
            *(uint4*)(dBl)     = make_uint4(loW[0], loW[1], loW[2], loW[3]);
            *(uint4*)(dBl + 4) = make_uint4(loW[4], loW[5], loW[6], loW[7]);
        }
    };

    // ---- Consumer geometry: wave (wm 0..1, wn 0..3) -> 64x32 region ----
    const int lane = tid & 31;
    const int wave = tid >> 5;
    const int lh   = lane & 15;
    const bool hiH = lane >= 16;
    const int wm   = wave >> 2;
    const int wn   = wave & 3;
    const int dB   = hiH ? 16 : 0;

    // Global bf16 B row pointers (PREB path): rows n0+lh and n0+16+lh.
    const unsigned *pBh0 = nullptr, *pBh1 = nullptr, *pBl0 = nullptr, *pBl1 = nullptr;
    if constexpr (PREB) {
        const size_t r0 = ((size_t)(b * Sn + nBlock + wn * 32 + lh)) * Dn;
        const size_t r1 = r0 + (size_t)16 * Dn;
        pBh0 = dh + r0 / 2; pBh1 = dh + r1 / 2;
        pBl0 = dl + r0 / 2; pBl1 = dl + r1 / 2;
    }

    v8f acc[4][2] = {};

    __syncthreads();  // sU ready
    produce(0, 0);
    __syncthreads();

    int cur = 0;
    for (int kk = 0; kk < Dn; kk += 32) {
        if (kk + 32 < Dn) produce(kk + 32, cur ^ 1);

        v16bf bhi[2], blo[2];
        if constexpr (PREB) {
            const int kw = (kk + dB) >> 1;  // dword offset into packed bf16 row
            bhi[0] = ldFrag(pBh0 + kw);
            bhi[1] = ldFrag(pBh1 + kw);
            blo[0] = ldFrag(pBl0 + kw);
            blo[1] = ldFrag(pBl1 + kw);
        } else {
            #pragma unroll
            for (int ni = 0; ni < 2; ++ni) {
                const int nt = wn * 2 + ni;
                bhi[ni] = ldFrag(sBhi + cur * FRAG_WORDS + (nt * 32 + lane) * 8);
                blo[ni] = ldFrag(sBlo + cur * FRAG_WORDS + (nt * 32 + lane) * 8);
            }
        }
        #pragma unroll
        for (int mi = 0; mi < 4; ++mi) {
            const int mt = wm * 4 + mi;
            const v16bf ahi = ldFrag(sAhi + cur * FRAG_WORDS + (mt * 32 + lane) * 8);
            const v16bf alo = ldFrag(sAlo + cur * FRAG_WORDS + (mt * 32 + lane) * 8);
            #pragma unroll
            for (int ni = 0; ni < 2; ++ni) {
                v8f c = acc[mi][ni];
                c = __builtin_amdgcn_wmma_f32_16x16x32_bf16(false, ahi, false, bhi[ni], (short)0, c, false, false);
                c = __builtin_amdgcn_wmma_f32_16x16x32_bf16(false, ahi, false, blo[ni], (short)0, c, false, false);
                c = __builtin_amdgcn_wmma_f32_16x16x32_bf16(false, alo, false, bhi[ni], (short)0, c, false, false);
                acc[mi][ni] = c;
            }
        }
        __syncthreads();
        cur ^= 1;
    }

    // ---- Epilogue: + t2_h[row] + t2_d[col] + bias[l] ----
    const float bl = bias[l];
    const size_t outBase = (size_t)z * Sn * Sn;
    const int rOff = hiH ? 8 : 0;
    const int m0 = mBlock + wm * 64;
    const int n0 = nBlock + wn * 32;
    #pragma unroll
    for (int mi = 0; mi < 4; ++mi) {
        #pragma unroll
        for (int ni = 0; ni < 2; ++ni) {
            const int col = n0 + ni * 16 + lh;
            const float td = sTd[col - nBlock] + bl;
            #pragma unroll
            for (int r = 0; r < 8; ++r) {
                const int row = m0 + mi * 16 + r + rOff;
                out[outBase + (size_t)row * Sn + col] = acc[mi][ni][r] + sTh[row - mBlock] + td;
            }
        }
    }
}

extern "C" void kernel_launch(void* const* d_in, const int* in_sizes, int n_in,
                              void* d_out, int out_size, void* d_ws, size_t ws_size,
                              hipStream_t stream) {
    (void)in_sizes; (void)n_in; (void)out_size;
    const float* head = (const float*)d_in[0];
    const float* dep  = (const float*)d_in[1];
    const float* U    = (const float*)d_in[2];
    const float* W    = (const float*)d_in[3];
    const float* bias = (const float*)d_in[4];
    float* out = (float*)d_out;

    dim3 grid(Sn / 128, Sn / 128, Bn * Ln);  // 4 x 4 x 256

    if (d_ws != nullptr && ws_size >= WS_NEED) {
        unsigned* dh = (unsigned*)d_ws;
        unsigned* dl = dh + HALF_WORDS;
        split_dep_kernel<<<(Bn * Sn * Dn / 8) / 256, 256, 0, stream>>>(dep, dh, dl);
        hipFuncSetAttribute(reinterpret_cast<const void*>(&biaffine_wmma_kernel<true>),
                            hipFuncAttributeMaxDynamicSharedMemorySize, (int)SMEM_PRE);
        biaffine_wmma_kernel<true><<<grid, 256, SMEM_PRE, stream>>>(
            head, dep, U, W, bias, out, dh, dl);
    } else {
        hipFuncSetAttribute(reinterpret_cast<const void*>(&biaffine_wmma_kernel<false>),
                            hipFuncAttributeMaxDynamicSharedMemorySize, (int)SMEM_ALL);
        biaffine_wmma_kernel<false><<<grid, 256, SMEM_ALL, stream>>>(
            head, dep, U, W, bias, out, nullptr, nullptr);
    }
}
